// Large_SNN_37134287242089
// MI455X (gfx1250) — matrix operand order repaired and checked by reading, not compiled, not run on hardware
//
#include <hip/hip_runtime.h>
#include <hip/hip_bf16.h>

// ---------------------------------------------------------------------------
// Spiking CNN (DQN trunk + LIF) for MI455X / gfx1250.
// - All GEMM-like layers on v_wmma_f32_16x16x32_f16 (wave32 WMMA).
// - conv2/conv3 stage per-image spike maps into LDS via the CDNA5 async
//   global->LDS path (global_load_async_to_lds_b128, ASYNCcnt), then do the
//   im2col gathers from LDS (ds_load) instead of 2-byte global gathers.
// - f32 membrane state, f16 spikes (exact {0,1}).
// Workspace: ~71 MB (L2-resident on the 192MB L2).
// ---------------------------------------------------------------------------

typedef __attribute__((ext_vector_type(16))) _Float16 v16h;
typedef __attribute__((ext_vector_type(8)))  _Float16 v8h;
typedef __attribute__((ext_vector_type(8)))  float    v8f;

#define SNN_BETA   0.95f
#define SNN_THRESH 1.0f

// ---- workspace layout (bytes) ---------------------------------------------
#define OFF_W1H   0u            // 32*64   f16 =     4096
#define OFF_W2H   4096u         // 64*512  f16 =    65536
#define OFF_W3H   69632u        // 64*576  f16 =    73728
#define OFF_WLH   143360u       // 512*3136 f16 = 3211264
#define OFF_MEM1  3354624u      // 512*32*400 f32 = 26214400
#define OFF_MEM2  29569024u     // 512*64*81  f32 = 10616832
#define OFF_MEM3  40185856u     // 512*64*49  f32 =  6422528
#define OFF_MEMF  46608384u     // 512*512    f32 =  1048576
#define OFF_ACC   47656960u     // 512*512    f32 =  1048576
#define OFF_S1    48705536u     // 512*32*400 f16 = 13107200
#define OFF_S2    61812736u     // 512*64*81  f16 =  5308416
#define OFF_S3    67121152u     // 512*3136   f16 =  3211264
#define MEM_ZERO_OFF   OFF_MEM1
#define MEM_ZERO_BYTES (OFF_S1 - OFF_MEM1)   // mem1..acc contiguous

// ---------------------------------------------------------------------------
// Async global->LDS b128 copy (CDNA5 VGLOBAL opcode 98, tracked by ASYNCcnt).
// lds_off: LDS byte offset (low 32 bits of the generic __shared__ address;
// ISA aperture mapping truncates generic LDS addresses to addr[31:0]).
// ---------------------------------------------------------------------------
static __device__ __forceinline__
void async_copy_b128_to_lds(unsigned lds_off, const void* gptr)
{
    asm volatile("global_load_async_to_lds_b128 %0, %1, off"
                 :: "v"(lds_off), "v"(gptr)
                 : "memory");
}
static __device__ __forceinline__ void wait_asynccnt0()
{
    asm volatile("s_wait_asynccnt 0" ::: "memory");
}

// ---------------------------------------------------------------------------
// A-fragment loader: 16-bit A 16x32 in this file's contiguous k order (not the ISA table order):
//   lane l: row M = l%16 ; half = l/16
//   v16h elems [0..7]  = K = kstep*32 + half*16     + 0..7   (contiguous)
//   v16h elems [8..15] = K = kstep*32 + half*16 + 8 + 0..7   (contiguous)
// -> two 16B (b128) loads from a row-major [M][K] f16 weight matrix.
// ---------------------------------------------------------------------------
static __device__ __forceinline__
v16h load_a_frag(const _Float16* __restrict__ w, int row, int K, int kstep, int half)
{
    const _Float16* base = w + (size_t)row * K + kstep * 32 + half * 16;
    v8h lo = *(const v8h*)(base);
    v8h hi = *(const v8h*)(base + 8);
    v16h a;
#pragma unroll
    for (int i = 0; i < 8; ++i) { a[i] = lo[i]; a[8 + i] = hi[i]; }
    return a;
}

// LIF (snntorch Leaky, reset_mechanism='subtract'), f32 state
static __device__ __forceinline__
float lif_update(float* __restrict__ mem, size_t idx, float inp)
{
    float m     = mem[idx];
    float reset = (m > SNN_THRESH) ? 1.0f : 0.0f;
    m = fmaf(SNN_BETA, m, inp) - reset * SNN_THRESH;
    mem[idx] = m;
    return (m > SNN_THRESH) ? 1.0f : 0.0f;
}

// ---------------------------------------------------------------------------
// conv1 + LIF1 : x[t]/255 -> 32x20x20.  GEMM M=32,K=64,N=512*400.
// grid 3200x128 -> 12800 waves = 512 imgs * 25 full N-tiles (EXEC all ones).
// x patch rows are contiguous in memory -> compiler emits b128 f32 loads.
// ---------------------------------------------------------------------------
__global__ __launch_bounds__(128)
void conv1_lif_kernel(const float* __restrict__ x, const _Float16* __restrict__ w1h,
                      const float* __restrict__ b1, float* __restrict__ mem1,
                      _Float16* __restrict__ s1, int t)
{
    const int lane  = threadIdx.x & 31;
    const int wid   = (blockIdx.x * 128 + threadIdx.x) >> 5;
    const int img   = wid / 25;
    const int ptile = wid % 25;
    const int half  = lane >> 4;
    const int lm    = lane & 15;
    const int p  = ptile * 16 + lm;          // output position 0..399, always valid
    const int oy = p / 20, ox = p % 20;
    const float* xin = x + ((size_t)img * 4 + t) * (84 * 84);

    v8f c0 = {}; v8f c1 = {};
#pragma unroll
    for (int ks = 0; ks < 2; ++ks) {
        // B fragment: B 32x16, contiguous k order matching A: lane column N=l%16, K = half*16+0..15
        v16h bf;
        const int kbase = ks * 32 + half * 16;
#pragma unroll
        for (int j = 0; j < 16; ++j) {
            const int K  = kbase + j;
            const int ky = K >> 3, kx = K & 7;
            bf[j] = (_Float16)(xin[(oy * 4 + ky) * 84 + (ox * 4 + kx)] * (1.0f / 255.0f));
        }
        v16h a0 = load_a_frag(w1h, lm,      64, ks, half);
        v16h a1 = load_a_frag(w1h, lm + 16, 64, ks, half);
        c0 = __builtin_amdgcn_wmma_f32_16x16x32_f16(false, a0, false, bf, (short)0, c0, false, false);
        c1 = __builtin_amdgcn_wmma_f32_16x16x32_f16(false, a1, false, bf, (short)0, c1, false, false);
    }
    // epilogue: C layout VGPR r -> M = half*8 + r, N = l%16
#pragma unroll
    for (int mt = 0; mt < 2; ++mt) {
        v8f c = mt ? c1 : c0;
#pragma unroll
        for (int r = 0; r < 8; ++r) {
            const int ch  = mt * 16 + half * 8 + r;
            const size_t idx = ((size_t)img * 32 + ch) * 400 + p;
            const float spk = lif_update(mem1, idx, c[r] + b1[ch]);
            s1[idx] = (_Float16)spk;
        }
    }
}

// ---------------------------------------------------------------------------
// conv2 + LIF2 : s1 -> 64x9x9.  GEMM M=64,K=512,N=512*81.
// One image per block; 192 threads = 6 waves = 6 N-tiles (last partial:
// reads clamped, writes predicated AFTER all wmma so EXEC stays all-ones).
// s1 image (25.6 KB) staged in LDS via async b128 copies, gathered with ds.
// ---------------------------------------------------------------------------
__global__ __launch_bounds__(192)
void conv2_lif_kernel(const _Float16* __restrict__ s1, const _Float16* __restrict__ w2h,
                      const float* __restrict__ b2, float* __restrict__ mem2,
                      _Float16* __restrict__ s2)
{
    __shared__ __align__(16) _Float16 tile[32 * 400];      // 25600 B
    const int tid = threadIdx.x;
    const int img = blockIdx.x;

    {   // stage s1[img] -> LDS: 1600 async b128 chunks across 192 threads
        const char* gbase = (const char*)(s1 + (size_t)img * 12800);
        const unsigned lbase = (unsigned)(size_t)(void*)tile;
        for (int i = tid; i < 1600; i += 192)
            async_copy_b128_to_lds(lbase + i * 16, gbase + i * 16);
        wait_asynccnt0();
    }
    __syncthreads();

    const int lane  = tid & 31;
    const int ptile = tid >> 5;                     // wave id 0..5
    const int half  = lane >> 4;
    const int lm    = lane & 15;
    const int p  = ptile * 16 + lm;                 // 0..95 (81 valid)
    const int pc = (p < 81) ? p : 80;               // clamp for reads
    const int oy = pc / 9, ox = pc % 9;

    v8f c[4] = {};
#pragma unroll 4
    for (int ks = 0; ks < 16; ++ks) {
        v16h bf;
        const int kbase = ks * 32 + half * 16;
#pragma unroll
        for (int j = 0; j < 16; ++j) {
            const int K  = kbase + j;
            const int ci = K >> 4;
            const int kk = K & 15;
            const int ky = kk >> 2, kx = kk & 3;
            bf[j] = tile[ci * 400 + (oy * 2 + ky) * 20 + (ox * 2 + kx)];
        }
#pragma unroll
        for (int mt = 0; mt < 4; ++mt) {
            v16h a = load_a_frag(w2h, mt * 16 + lm, 512, ks, half);
            c[mt] = __builtin_amdgcn_wmma_f32_16x16x32_f16(false, a, false, bf, (short)0, c[mt], false, false);
        }
    }
    if (p < 81) {
#pragma unroll
        for (int mt = 0; mt < 4; ++mt)
#pragma unroll
            for (int r = 0; r < 8; ++r) {
                const int ch  = mt * 16 + half * 8 + r;
                const size_t idx = ((size_t)img * 64 + ch) * 81 + p;
                const float spk = lif_update(mem2, idx, c[mt][r] + b2[ch]);
                s2[idx] = (_Float16)spk;
            }
    }
}

// ---------------------------------------------------------------------------
// conv3 + LIF3 : s2 -> 64x7x7.  GEMM M=64,K=576,N=512*49.
// One image per block; 128 threads = 4 waves = 4 N-tiles (last partial).
// s2 image (10.4 KB) staged in LDS via async b128 copies.
// ---------------------------------------------------------------------------
__global__ __launch_bounds__(128)
void conv3_lif_kernel(const _Float16* __restrict__ s2, const _Float16* __restrict__ w3h,
                      const float* __restrict__ b3, float* __restrict__ mem3,
                      _Float16* __restrict__ s3)
{
    __shared__ __align__(16) _Float16 tile[64 * 81];       // 10368 B
    const int tid = threadIdx.x;
    const int img = blockIdx.x;

    {   // stage s2[img] -> LDS: 648 async b128 chunks across 128 threads
        const char* gbase = (const char*)(s2 + (size_t)img * 5184);
        const unsigned lbase = (unsigned)(size_t)(void*)tile;
        for (int i = tid; i < 648; i += 128)
            async_copy_b128_to_lds(lbase + i * 16, gbase + i * 16);
        wait_asynccnt0();
    }
    __syncthreads();

    const int lane  = tid & 31;
    const int ptile = tid >> 5;                     // wave id 0..3
    const int half  = lane >> 4;
    const int lm    = lane & 15;
    const int p  = ptile * 16 + lm;                 // 0..63 (49 valid)
    const int pc = (p < 49) ? p : 48;
    const int oy = pc / 7, ox = pc % 7;

    v8f c[4] = {};
#pragma unroll 3
    for (int ks = 0; ks < 18; ++ks) {               // K = 576 = 18*32
        v16h bf;
        const int kbase = ks * 32 + half * 16;
#pragma unroll
        for (int j = 0; j < 16; ++j) {
            const int K  = kbase + j;
            const int ci = K / 9;
            const int kk = K % 9;
            const int ky = kk / 3, kx = kk % 3;
            bf[j] = tile[ci * 81 + (oy + ky) * 9 + (ox + kx)];
        }
#pragma unroll
        for (int mt = 0; mt < 4; ++mt) {
            v16h a = load_a_frag(w3h, mt * 16 + lm, 576, ks, half);
            c[mt] = __builtin_amdgcn_wmma_f32_16x16x32_f16(false, a, false, bf, (short)0, c[mt], false, false);
        }
    }
    if (p < 49) {
#pragma unroll
        for (int mt = 0; mt < 4; ++mt)
#pragma unroll
            for (int r = 0; r < 8; ++r) {
                const int ch  = mt * 16 + half * 8 + r;
                // s3 stored as [B][3136] row-major: ch*49 + p == flatten(C-major)
                const size_t idx = ((size_t)img * 64 + ch) * 49 + p;
                const float spk = lif_update(mem3, idx, c[mt][r] + b3[ch]);
                s3[idx] = (_Float16)spk;
            }
    }
}

// ---------------------------------------------------------------------------
// FC 3136->512 + LIF + spike accumulate.  GEMM M=512,K=3136,N=512(batch).
// grid 64x128 -> 256 waves = 8 M-groups(64 rows) x 32 N-tiles. B fragment is
// a single contiguous 32B-aligned v16h load (C-major flatten of s3).
// ---------------------------------------------------------------------------
__global__ __launch_bounds__(128)
void fc_lif_kernel(const _Float16* __restrict__ s3, const _Float16* __restrict__ wlh,
                   const float* __restrict__ bl, float* __restrict__ memf,
                   float* __restrict__ acc)
{
    const int lane = threadIdx.x & 31;
    const int wid  = (blockIdx.x * 128 + threadIdx.x) >> 5;
    const int mg   = wid >> 5;                      // 0..7  (64 output rows each)
    const int nt   = wid & 31;                      // 0..31 (16 batch cols each)
    const int half = lane >> 4;
    const int lm   = lane & 15;
    const int b    = nt * 16 + lm;                  // batch index (column)
    const _Float16* bp = s3 + (size_t)b * 3136;

    v8f c[4] = {};
#pragma unroll 2
    for (int ks = 0; ks < 98; ++ks) {               // K = 3136 = 98*32
        v16h bf = *(const v16h*)(bp + ks * 32 + half * 16);   // 32B aligned
#pragma unroll
        for (int mt = 0; mt < 4; ++mt) {
            v16h a = load_a_frag(wlh, mg * 64 + mt * 16 + lm, 3136, ks, half);
            c[mt] = __builtin_amdgcn_wmma_f32_16x16x32_f16(false, a, false, bf, (short)0, c[mt], false, false);
        }
    }
#pragma unroll
    for (int mt = 0; mt < 4; ++mt)
#pragma unroll
        for (int r = 0; r < 8; ++r) {
            const int o = mg * 64 + mt * 16 + half * 8 + r;
            const size_t idx = (size_t)b * 512 + o;
            const float spk = lif_update(memf, idx, c[mt][r] + bl[o]);
            acc[idx] += spk;
        }
}

// ---------------------------------------------------------------------------
// Head: out[b][j] = (acc[b]/4) . wh[j] + bh[j]   (512x512x18, f32 VALU)
// ---------------------------------------------------------------------------
__global__ __launch_bounds__(128)
void head_kernel(const float* __restrict__ acc, const float* __restrict__ wh,
                 const float* __restrict__ bh, float* __restrict__ out)
{
    const int i = blockIdx.x * 128 + threadIdx.x;   // 0..9215
    const int b = i / 18, j = i % 18;
    const float* ap = acc + (size_t)b * 512;
    const float* wp = wh + (size_t)j * 512;
    float s = 0.0f;
#pragma unroll 8
    for (int o = 0; o < 512; ++o) s = fmaf(ap[o], wp[o], s);
    out[i] = s * 0.25f + bh[j];
}

__global__ __launch_bounds__(128)
void cvt_f32_to_f16_kernel(const float* __restrict__ src, _Float16* __restrict__ dst, int n)
{
    const int i = blockIdx.x * 128 + threadIdx.x;
    if (i < n) dst[i] = (_Float16)src[i];
}

// ---------------------------------------------------------------------------
extern "C" void kernel_launch(void* const* d_in, const int* in_sizes, int n_in,
                              void* d_out, int out_size, void* d_ws, size_t ws_size,
                              hipStream_t stream)
{
    (void)in_sizes; (void)n_in; (void)out_size; (void)ws_size;
    const float* x  = (const float*)d_in[0];
    const float* w1 = (const float*)d_in[1];
    const float* b1 = (const float*)d_in[2];
    const float* w2 = (const float*)d_in[3];
    const float* b2 = (const float*)d_in[4];
    const float* w3 = (const float*)d_in[5];
    const float* b3 = (const float*)d_in[6];
    const float* wl = (const float*)d_in[7];
    const float* bl = (const float*)d_in[8];
    const float* wh = (const float*)d_in[9];
    const float* bh = (const float*)d_in[10];

    char* ws = (char*)d_ws;
    _Float16* w1h  = (_Float16*)(ws + OFF_W1H);
    _Float16* w2h  = (_Float16*)(ws + OFF_W2H);
    _Float16* w3h  = (_Float16*)(ws + OFF_W3H);
    _Float16* wlh  = (_Float16*)(ws + OFF_WLH);
    float*    mem1 = (float*)(ws + OFF_MEM1);
    float*    mem2 = (float*)(ws + OFF_MEM2);
    float*    mem3 = (float*)(ws + OFF_MEM3);
    float*    memf = (float*)(ws + OFF_MEMF);
    float*    accb = (float*)(ws + OFF_ACC);
    _Float16* s1   = (_Float16*)(ws + OFF_S1);
    _Float16* s2   = (_Float16*)(ws + OFF_S2);
    _Float16* s3   = (_Float16*)(ws + OFF_S3);

    // weights -> f16 (deterministic, every launch)
    cvt_f32_to_f16_kernel<<<(2048 + 127) / 128, 128, 0, stream>>>(w1, w1h, 2048);
    cvt_f32_to_f16_kernel<<<(32768 + 127) / 128, 128, 0, stream>>>(w2, w2h, 32768);
    cvt_f32_to_f16_kernel<<<(36864 + 127) / 128, 128, 0, stream>>>(w3, w3h, 36864);
    cvt_f32_to_f16_kernel<<<(1605632 + 127) / 128, 128, 0, stream>>>(wl, wlh, 1605632);

    // zero membrane state + spike accumulator (graph-capturable memset node)
    hipMemsetAsync(ws + MEM_ZERO_OFF, 0, MEM_ZERO_BYTES, stream);

    for (int t = 0; t < 4; ++t) {
        conv1_lif_kernel<<<3200, 128, 0, stream>>>(x, w1h, b1, mem1, s1, t);
        conv2_lif_kernel<<<512, 192, 0, stream>>>(s1, w2h, b2, mem2, s2);
        conv3_lif_kernel<<<512, 128, 0, stream>>>(s2, w3h, b3, mem3, s3);
        fc_lif_kernel<<<64, 128, 0, stream>>>(s3, wlh, bl, memf, accb);
    }
    head_kernel<<<72, 128, 0, stream>>>(accb, wh, bh, (float*)d_out);
}
